// sparse_conv_9148280340890
// MI455X (gfx1250) — matrix operand order repaired and checked by reading, not compiled, not run on hardware
//
#include <hip/hip_runtime.h>

// ---------------------------------------------------------------------------
// MI455X (gfx1250, wave32) sparse-block conv pipeline.
// All GEMMs use V_WMMA_F32_16X16X32_BF16 (bf16 inputs, f32 accumulate):
// fp32 matrix K=4 would leave the 309-GFLOP 3x3 conv at >1ms; bf16 WMMA
// brings compute near the ~50us HBM roofline for ~800MB of traffic.
// Activation tiles are staged into LDS with GLOBAL_LOAD_ASYNC_TO_LDS_B128
// (ASYNCcnt-tracked, no VGPR bounce) — the CDNA5 async-copy path.
// ---------------------------------------------------------------------------

typedef __bf16 bf16;
typedef __attribute__((ext_vector_type(16))) __bf16 v16bf;
typedef __attribute__((ext_vector_type(8)))  __bf16 v8bf;
typedef __attribute__((ext_vector_type(8)))  float  v8f;

#define HW 65536   // 256*256

__device__ __forceinline__ bf16 f2bf(float f) { return (bf16)f; }

union U16bf { uint4 q[2]; v16bf v; };
union U8bf  { uint4 q;    v8bf  v; };

// CDNA5 async global->LDS copy, 16 bytes per active lane. VDST holds the
// LDS byte offset (low 32 bits of the shared-aperture flat address).
__device__ __forceinline__ void async_copy16(const void* gptr, void* lptr) {
  unsigned lds_off = (unsigned)(uintptr_t)lptr;
  asm volatile("global_load_async_to_lds_b128 %0, %1, off"
               :: "v"(lds_off), "v"(gptr)
               : "memory");
}
__device__ __forceinline__ void wait_async0() {
  asm volatile("s_wait_asynccnt 0x0" ::: "memory");
}

// ---------------------------------------------------------------------------
// Fused conv (1x1 or 3x3, per 64-pixel group) + bias + ReLU + BN-stats.
//   Input: pixel-major bf16 [group][64][CIN]   (or raw NCHW f32 for stage 0)
//   Weights: bf16 [KS*KS][COUT][CIN] row-major
//   Output: pixel-major bf16 [group][64][COUT], plus per-channel sum/sumsq.
// 256 threads = 8 waves. Wave w owns M-tiles {w, w+8}, all 4 N-tiles (N=64).
// ---------------------------------------------------------------------------
template<int CIN, int COUT, int KS, bool STAGE0>
__global__ __launch_bounds__(256) void conv_bn_stats(
    const bf16*  __restrict__ gin,
    const float* __restrict__ x,
    const bf16*  __restrict__ W,
    const float* __restrict__ bias,
    bf16*        __restrict__ gout,
    float*       __restrict__ gsum,
    float*       __restrict__ gsumsq)
{
  constexpr int MT  = COUT / 16;   // M tiles
  constexpr int MPW = MT / 8;      // M tiles per wave (1 or 2)

  __shared__ alignas(32) bf16 ldsIn[64 * CIN];   // pixel-major [p][ci]
  __shared__ float ldsS[COUT];
  __shared__ float ldsQ[COUT];

  const int tid  = threadIdx.x;
  const int grp  = blockIdx.x;
  const int wave = tid >> 5;
  const int lane = tid & 31;
  const int hl   = lane >> 4;   // which 16-lane half
  const int col  = lane & 15;

  for (int c = tid; c < COUT; c += 256) { ldsS[c] = 0.f; ldsQ[c] = 0.f; }

  if constexpr (STAGE0) {
    // group = 64 consecutive hw of one image; transpose NCHW -> [p][ci] here
    const int n   = grp >> 10;
    const int hw0 = (grp & 1023) << 6;
    const float* xb = x + (size_t)n * CIN * HW + hw0;
    for (int i = tid; i < 64 * CIN; i += 256) {
      int c = i >> 6, p = i & 63;                 // coalesced 64-float rows
      ldsIn[p * CIN + c] = f2bf(xb[(size_t)c * HW + p]);
    }
  } else {
    // contiguous 64*CIN*2-byte tile: async DMA straight into LDS
    const char* src = (const char*)(gin + (size_t)grp * 64 * CIN);
    char* dst = (char*)ldsIn;
    for (int i = tid; i < 64 * CIN / 8; i += 256)
      async_copy16(src + (size_t)i * 16, dst + (size_t)i * 16);
    wait_async0();
  }
  __syncthreads();

  v8f acc[MPW * 4];
  #pragma unroll
  for (int i = 0; i < MPW * 4; ++i) acc[i] = (v8f)0.0f;

  const v16bf bzero = (v16bf)(bf16)0.0f;

  for (int t = 0; t < KS * KS; ++t) {              // 3x3 taps (or single)
    int qpix[4]; bool ok[4];
    #pragma unroll
    for (int nt = 0; nt < 4; ++nt) {
      int p = nt * 16 + col;
      if constexpr (KS == 3) {
        int yy = (p >> 3) + (t / 3) - 1;
        int xx = (p & 7)  + (t % 3) - 1;
        ok[nt]   = ((unsigned)yy < 8u) && ((unsigned)xx < 8u);  // zero pad
        qpix[nt] = (yy << 3) + xx;
      } else { ok[nt] = true; qpix[nt] = p; }
    }
    for (int ci0 = 0; ci0 < CIN; ci0 += 32) {      // K step of 32
      // B fragments: N=pixel (lane col), K = ci0 + 16*hl .. +15, contiguous
      v16bf bfrag[4];
      #pragma unroll
      for (int nt = 0; nt < 4; ++nt)
        bfrag[nt] = ok[nt] ? *(const v16bf*)&ldsIn[qpix[nt] * CIN + ci0 + hl * 16]
                           : bzero;
      #pragma unroll
      for (int mi = 0; mi < MPW; ++mi) {
        // A fragment: M = lane col row; contiguous K = ci0 + 16*hl .. +15, same as B
        const int m = (wave + mi * 8) * 16 + col;
        U16bf a;
        const uint4* ap = (const uint4*)(W + ((size_t)t * COUT + m) * CIN + ci0 + hl * 16);
        a.q[0] = ap[0];   // K = ci0+16*hl .. +7
        a.q[1] = ap[1];   // K = ci0+16*hl+8 .. +7
        #pragma unroll
        for (int nt = 0; nt < 4; ++nt)
          acc[mi * 4 + nt] = __builtin_amdgcn_wmma_f32_16x16x32_bf16(
              false, a.v, false, bfrag[nt], (short)0, acc[mi * 4 + nt], false, false);
      }
    }
  }

  // Epilogue: bias + ReLU, stats into LDS, bf16 store (8 consecutive cout)
  #pragma unroll
  for (int mi = 0; mi < MPW; ++mi) {
    const int mb = (wave + mi * 8) * 16 + hl * 8;
    #pragma unroll
    for (int nt = 0; nt < 4; ++nt) {
      const int p = nt * 16 + col;
      U8bf ob;
      v8f v = acc[mi * 4 + nt];
      #pragma unroll
      for (int j = 0; j < 8; ++j) {
        float f = v[j] + bias[mb + j];
        f = fmaxf(f, 0.0f);
        atomicAdd(&ldsS[mb + j], f);
        atomicAdd(&ldsQ[mb + j], f * f);
        ob.v[j] = f2bf(f);
      }
      *(uint4*)&gout[((size_t)grp * 64 + p) * COUT + mb] = ob.q;
    }
  }
  __syncthreads();
  for (int c = tid; c < COUT; c += 256) {
    atomicAdd(&gsum[c],   ldsS[c]);
    atomicAdd(&gsumsq[c], ldsQ[c]);
  }
}

// ---------------------------------------------------------------------------
__global__ void bn_finalize(const float* __restrict__ sum, const float* __restrict__ sumsq,
                            const float* __restrict__ gamma, const float* __restrict__ beta,
                            float* __restrict__ scale, float* __restrict__ shift,
                            int C, float invN)
{
  int c = blockIdx.x * blockDim.x + threadIdx.x;
  if (c < C) {
    float m   = sum[c] * invN;
    float var = fmaxf(sumsq[c] * invN - m * m, 0.0f);
    float s   = gamma[c] * rsqrtf(var + 1e-5f);
    scale[c] = s;
    shift[c] = beta[c] - m * s;
  }
}

__global__ void zero_f32(float* p, int n) {
  int i = blockIdx.x * blockDim.x + threadIdx.x;
  if (i < n) p[i] = 0.f;
}

// fp32 weights -> bf16; w2 rearranged [co][ci][3][3] -> [tap][co][ci]
__global__ void convert_weights(const float* __restrict__ wc, const float* __restrict__ w1,
                                const float* __restrict__ w2, const float* __restrict__ w3,
                                bf16* __restrict__ wcb, bf16* __restrict__ w1b,
                                bf16* __restrict__ w2b, bf16* __restrict__ w3b)
{
  int i = blockIdx.x * blockDim.x + threadIdx.x;
  if (i < 128 * 64)  wcb[i] = f2bf(wc[i]);
  if (i < 256 * 128) { w1b[i] = f2bf(w1[i]); w3b[i] = f2bf(w3[i]); }
  if (i < 256 * 256 * 9) {
    int co = i / 2304;
    int r  = i % 2304;
    int ci = r / 9;
    int t  = r % 9;
    w2b[((size_t)t * 256 + co) * 256 + ci] = f2bf(w2[i]);
  }
}

// Whole-tensor BN0 apply: pixel-major bf16 -> NCHW f32 d_out, LDS-transposed
// so both the 256B reads and 512B row writes stay coalesced (268MB stream).
__global__ __launch_bounds__(256) void scatter_full_k(const bf16* __restrict__ y0,
    const float* __restrict__ scale, const float* __restrict__ shift,
    float* __restrict__ out)
{
  __shared__ alignas(16) bf16 tile[128 * 128];   // [c][hw]
  int n   = blockIdx.x >> 9;
  int hw0 = (blockIdx.x & 511) << 7;
  const uint4* src = (const uint4*)(y0 + ((size_t)n * HW + hw0) * 128);
  for (int i = threadIdx.x; i < 2048; i += 256) {
    int r  = i >> 4;
    int cg = (i & 15) * 8;
    U8bf v; v.q = src[(size_t)r * 16 + (i & 15)];
    #pragma unroll
    for (int j = 0; j < 8; ++j) tile[(cg + j) * 128 + r] = v.v[j];
  }
  __syncthreads();
  int c  = threadIdx.x >> 1;
  int h0 = (threadIdx.x & 1) << 6;
  float s = scale[c], sh = shift[c];
  float4* dst = (float4*)(out + (((size_t)n * 128 + c) << 16) + hw0 + h0);
  #pragma unroll
  for (int j = 0; j < 16; ++j) {
    float4 o;
    o.x = s * (float)tile[c * 128 + h0 + j * 4 + 0] + sh;
    o.y = s * (float)tile[c * 128 + h0 + j * 4 + 1] + sh;
    o.z = s * (float)tile[c * 128 + h0 + j * 4 + 2] + sh;
    o.w = s * (float)tile[c * 128 + h0 + j * 4 + 3] + sh;
    dst[j] = o;
  }
}

// Gather active 8x8 blocks of BN0 output -> g0 pixel-major bf16
__global__ void gather_k(const bf16* __restrict__ y0, const int* __restrict__ abi,
                         const float* __restrict__ scale, const float* __restrict__ shift,
                         bf16* __restrict__ g0, int nb)
{
  int i = blockIdx.x * blockDim.x + threadIdx.x;
  if (i >= nb * 1024) return;
  int b  = i >> 10;
  int p  = (i >> 4) & 63;
  int cg = (i & 15) * 8;
  int n = abi[b * 3 + 0], bi = abi[b * 3 + 1], bj = abi[b * 3 + 2];
  int hw = ((bi * 8 + (p >> 3)) << 8) + bj * 8 + (p & 7);
  U8bf v; v.q = *(const uint4*)(y0 + ((size_t)n * HW + hw) * 128 + cg);
  U8bf o;
  #pragma unroll
  for (int j = 0; j < 8; ++j)
    o.v[j] = f2bf(scale[cg + j] * (float)v.v[j] + shift[cg + j]);
  *(uint4*)(g0 + (size_t)i * 8) = o.q;
}

// In-place BN apply on a pixel-major bf16 tensor
__global__ void apply_bn_inplace(bf16* __restrict__ g, const float* __restrict__ scale,
                                 const float* __restrict__ shift, int Cdiv8, long total8)
{
  long i = (long)blockIdx.x * blockDim.x + threadIdx.x;
  if (i >= total8) return;
  int cb = (int)(i % Cdiv8) * 8;
  U8bf v; v.q = *(uint4*)(g + i * 8);
  U8bf o;
  #pragma unroll
  for (int j = 0; j < 8; ++j)
    o.v[j] = f2bf(scale[cb + j] * (float)v.v[j] + shift[cb + j]);
  *(uint4*)(g + i * 8) = o.q;
}

// Final BN3 apply + scatter of block results into NCHW f32 d_out
__global__ void scatter_blocks_k(const bf16* __restrict__ g3, const int* __restrict__ abi,
                                 const float* __restrict__ scale, const float* __restrict__ shift,
                                 float* __restrict__ out, int nb)
{
  int i = blockIdx.x * blockDim.x + threadIdx.x;
  if (i >= nb * 1024) return;
  int b  = i >> 10;
  int c  = i & 127;                 // consecutive lanes -> contiguous g3 reads
  int py = (i >> 7) & 7;
  int n = abi[b * 3 + 0], bi = abi[b * 3 + 1], bj = abi[b * 3 + 2];
  float s = scale[c], sh = shift[c];
  const bf16* src = g3 + ((size_t)b * 64 + py * 8) * 128 + c;
  float* dst = out + (((size_t)(n * 128 + c)) << 16) + ((bi * 8 + py) << 8) + (bj * 8);
  float4 o0, o1;
  o0.x = s * (float)src[0]   + sh; o0.y = s * (float)src[128] + sh;
  o0.z = s * (float)src[256] + sh; o0.w = s * (float)src[384] + sh;
  o1.x = s * (float)src[512] + sh; o1.y = s * (float)src[640] + sh;
  o1.z = s * (float)src[768] + sh; o1.w = s * (float)src[896] + sh;
  ((float4*)dst)[0] = o0;
  ((float4*)dst)[1] = o1;
}

// ---------------------------------------------------------------------------
extern "C" void kernel_launch(void* const* d_in, const int* in_sizes, int n_in,
                              void* d_out, int out_size, void* d_ws, size_t ws_size,
                              hipStream_t stream)
{
  const float* x    = (const float*)d_in[0];
  const int*   abi  = (const int*)  d_in[1];
  const float* w_c  = (const float*)d_in[2];
  const float* b_c  = (const float*)d_in[3];
  const float* g_c  = (const float*)d_in[4];
  const float* be_c = (const float*)d_in[5];
  const float* w1   = (const float*)d_in[6];
  const float* b1   = (const float*)d_in[7];
  const float* ga1  = (const float*)d_in[8];
  const float* be1  = (const float*)d_in[9];
  const float* w2   = (const float*)d_in[10];
  const float* b2   = (const float*)d_in[11];
  const float* ga2  = (const float*)d_in[12];
  const float* be2  = (const float*)d_in[13];
  const float* w3   = (const float*)d_in[14];
  const float* b3   = (const float*)d_in[15];
  const float* ga3  = (const float*)d_in[16];
  const float* be3  = (const float*)d_in[17];
  float* out = (float*)d_out;

  const int nb = in_sizes[1] / 3;   // 4096 active blocks

  // Workspace partition (256B aligned)
  char* ws = (char*)d_ws;
  size_t off = 0;
  auto take = [&](size_t bytes) -> char* {
    char* p = ws + off;
    off = (off + bytes + 255) & ~(size_t)255;
    return p;
  };
  bf16* y0pm = (bf16*)take((size_t)8 * HW * 128 * 2);       // BN0 input (pre-BN, post-relu)
  bf16* gb1  = (bf16*)take((size_t)nb * 64 * 256 * 2);      // conv1 out
  bf16* gb2  = (bf16*)take((size_t)nb * 64 * 256 * 2);      // conv2 out
  bf16* gb0  = (bf16*)take((size_t)nb * 64 * 128 * 2);      // g0, reused as g3
  bf16* wcb  = (bf16*)take((size_t)128 * 64 * 2);
  bf16* w1b  = (bf16*)take((size_t)256 * 128 * 2);
  bf16* w2b  = (bf16*)take((size_t)9 * 256 * 256 * 2);
  bf16* w3b  = (bf16*)take((size_t)128 * 256 * 2);
  float* st  = (float*)take((size_t)3072 * 4);
  bf16* gb3  = gb0;   // conv1 has fully consumed g0 before conv3 writes

  float *sum0 = st,        *q0 = st + 128;
  float *sum1 = st + 256,  *q1 = st + 512;
  float *sum2 = st + 768,  *q2 = st + 1024;
  float *sum3 = st + 1280, *q3 = st + 1408;
  float *sc0 = st + 1536, *sh0 = st + 1664;
  float *sc1 = st + 1792, *sh1 = st + 2048;
  float *sc2 = st + 2304, *sh2 = st + 2560;
  float *sc3 = st + 2816, *sh3 = st + 2944;

  const float invFull = 1.0f / (8.0f * (float)HW);
  const float invBlk  = 1.0f / ((float)nb * 64.0f);

  zero_f32<<<6, 256, 0, stream>>>(st, 1536);
  convert_weights<<<2304, 256, 0, stream>>>(w_c, w1, w2, w3, wcb, w1b, w2b, w3b);

  // Stage 0: full-image 1x1 conv 64->128 (+relu+stats), 8192 pixel groups
  conv_bn_stats<64, 128, 1, true><<<8192, 256, 0, stream>>>(
      nullptr, x, wcb, b_c, y0pm, sum0, q0);
  bn_finalize<<<1, 256, 0, stream>>>(sum0, q0, g_c, be_c, sc0, sh0, 128, invFull);
  scatter_full_k<<<4096, 256, 0, stream>>>(y0pm, sc0, sh0, out);
  gather_k<<<nb * 4, 256, 0, stream>>>(y0pm, abi, sc0, sh0, gb0, nb);

  // Stage 1: per-block 1x1 conv 128->256
  conv_bn_stats<128, 256, 1, false><<<nb, 256, 0, stream>>>(
      gb0, nullptr, w1b, b1, gb1, sum1, q1);
  bn_finalize<<<1, 256, 0, stream>>>(sum1, q1, ga1, be1, sc1, sh1, 256, invBlk);
  apply_bn_inplace<<<nb * 8, 256, 0, stream>>>(gb1, sc1, sh1, 32, (long)nb * 2048);

  // Stage 2: per-block 3x3 conv 256->256 (zero-padded within each block)
  conv_bn_stats<256, 256, 3, false><<<nb, 256, 0, stream>>>(
      gb1, nullptr, w2b, b2, gb2, sum2, q2);
  bn_finalize<<<1, 256, 0, stream>>>(sum2, q2, ga2, be2, sc2, sh2, 256, invBlk);
  apply_bn_inplace<<<nb * 8, 256, 0, stream>>>(gb2, sc2, sh2, 32, (long)nb * 2048);

  // Stage 3: per-block 1x1 conv 256->128, then BN3 + scatter into d_out
  conv_bn_stats<256, 128, 1, false><<<nb, 256, 0, stream>>>(
      gb2, nullptr, w3b, b3, gb3, sum3, q3);
  bn_finalize<<<1, 256, 0, stream>>>(sum3, q3, ga3, be3, sc3, sh3, 128, invBlk);
  scatter_blocks_k<<<nb * 4, 256, 0, stream>>>(gb3, abi, sc3, sh3, out, nb);

  (void)n_in; (void)out_size; (void)ws_size;
}
